// Net_49830210568755
// MI455X (gfx1250) — matrix operand order repaired and checked by reading, not compiled, not run on hardware
//
#include <hip/hip_runtime.h>
#include <hip/hip_bf16.h>
#include <hip/hip_fp16.h>

typedef __attribute__((ext_vector_type(16))) _Float16 v16h;
typedef __attribute__((ext_vector_type(8)))  _Float16 h8v;
typedef __attribute__((ext_vector_type(8)))  float    v8f;

#define T_LEN 2000
#define HID   320
#define HID2  640
#define HID4  1280
#define DIN_  120
#define DINP  128
#define VOC   10000
#define VOCP  10016
#define NBEAM 4
#define NSTEP 200

// ---------------- conversion kernels ----------------

__global__ void k_f32_to_f16_pad(const float* __restrict__ X, _Float16* __restrict__ Y,
                                 int M, int K, int Kp) {
  int idx = blockIdx.x * blockDim.x + threadIdx.x;
  if (idx >= M * Kp) return;
  int m = idx / Kp, k = idx % Kp;
  Y[idx] = (k < K) ? (_Float16)X[(size_t)m * K + k] : (_Float16)0.f;
}

// W: (N,K) f32 row-major  ->  Wt: (Kp,N) f16 (zero-padded K)
__global__ void k_transpose_f16(const float* __restrict__ W, _Float16* __restrict__ Wt,
                                int N, int K, int Kp) {
  int idx = blockIdx.x * blockDim.x + threadIdx.x;
  if (idx >= Kp * N) return;
  int k = idx / N, n = idx % N;
  Wt[idx] = (k < K) ? (_Float16)W[(size_t)n * K + k] : (_Float16)0.f;
}

__global__ void k_zero_f16(_Float16* p, int n) {
  int i = blockIdx.x * blockDim.x + threadIdx.x;
  if (i < n) p[i] = (_Float16)0.f;
}

// ---------------- WMMA GEMM, NB-way N-tile register blocking ----------------
// C[M,N] = act(A16[M,Kp] x Bt16[Kp,N] + bias)
// A row-major f16 (ld=Kp), Bt K-major f16 (ld=N), C f32 (ld=N).
// M%16==0, N%(16*NB)==0, Kp%32==0. One wave per (16 x 16*NB) output strip:
// the A fragment is loaded once per K-step and reused across NB WMMAs.
template <int NB>
__global__ void k_wmma_gemm_nb(const _Float16* __restrict__ A, const _Float16* __restrict__ Bt,
                               const float* __restrict__ bias, float* __restrict__ C,
                               int M, int N, int Kp, int ngroups, int act) {
  int wave = (int)((blockIdx.x * blockDim.x + threadIdx.x) >> 5);
  int lane = threadIdx.x & 31;
  int mtiles = M >> 4;
  int mt = wave / ngroups, ng = wave % ngroups;
  if (mt >= mtiles) return;

  int row   = (mt << 4) + (lane & 15);
  int khalf = lane >> 4;
  v8f zero = {};
  v8f acc[NB];
#pragma unroll
  for (int i = 0; i < NB; ++i) acc[i] = zero;
  int ktiles = Kp >> 5;

  const _Float16* arow = A + (size_t)row * Kp + khalf * 8;
  const _Float16* bcol = Bt + (size_t)(khalf * 8) * N + (size_t)(ng * NB) * 16 + (lane & 15);

  for (int kt = 0; kt < ktiles; ++kt) {
    h8v lo = *(const h8v*)(arow);
    h8v hi = *(const h8v*)(arow + 16);
    v16h a;
#pragma unroll
    for (int j = 0; j < 8; ++j) { a[j] = lo[j]; a[8 + j] = hi[j]; }
#pragma unroll
    for (int i = 0; i < NB; ++i) {
      v16h b;
#pragma unroll
      for (int j = 0; j < 8; ++j) { b[j] = bcol[(size_t)j * N + i * 16]; b[8 + j] = bcol[(size_t)(16 + j) * N + i * 16]; }
      acc[i] = __builtin_amdgcn_wmma_f32_16x16x32_f16(false, a, false, b,
                                                      (short)0, acc[i], false, false);
    }
    arow += 32;
    bcol += (size_t)32 * N;
  }

  int mbase = (mt << 4) + (khalf << 3);
  int nbase = (ng * NB) << 4;
#pragma unroll
  for (int i = 0; i < NB; ++i) {
    int n = nbase + (i << 4) + (lane & 15);
    float bv = bias ? bias[n] : 0.f;
#pragma unroll
    for (int r = 0; r < 8; ++r) {
      float v = acc[i][r] + bv;
      if (act) v = tanhf(v);
      C[(size_t)(mbase + r) * N + n] = v;
    }
  }
}

// ---------------- LSTM recurrent scan (one block per direction) ----------------
// Next timestep's xproj row (1280 f32) is staged into double-buffered LDS with
// GLOBAL_LOAD_ASYNC_TO_LDS_B64 (ASYNCcnt) while the current dot products run.
__global__ void k_lstm_scan(const float* __restrict__ xprojF, const float* __restrict__ xprojB,
                            const float* __restrict__ whh, float* __restrict__ hout) {
  int dir = blockIdx.x;
  const float* xproj = dir ? xprojB : xprojF;
  const float* w = whh + (size_t)dir * HID4 * HID;
  __shared__ float h_sh[HID];
  __shared__ float g_sh[2][HID];       // gg, go produced by p==1
  __shared__ float xbuf[2][HID4];      // double-buffered xproj staging
  int tid = threadIdx.x;               // 0..639
  int u = tid % HID;
  int p = tid / HID;                   // 0: gates i,f   1: gates g,o
  if (p == 0) h_sh[u] = 0.f;
  float c = 0.f;
  // prologue: load step 0 synchronously
  {
    int t0 = dir ? (T_LEN - 1) : 0;
    xbuf[0][2 * tid]     = xproj[(size_t)t0 * HID4 + 2 * tid];
    xbuf[0][2 * tid + 1] = xproj[(size_t)t0 * HID4 + 2 * tid + 1];
  }
  __syncthreads();
  const float* w0 = w + (size_t)(2 * p * HID + u) * HID;
  const float* w1 = w + (size_t)((2 * p + 1) * HID + u) * HID;
  for (int s = 0; s < T_LEN; ++s) {
    int t = dir ? (T_LEN - 1 - s) : s;
    int cur = s & 1, nxt = cur ^ 1;
    if (s + 1 < T_LEN) {
      // async copy next xproj row into the other LDS buffer (8B per lane)
      int tn = dir ? (T_LEN - 2 - s) : (s + 1);
      unsigned int ldsa = (unsigned int)(size_t)&xbuf[nxt][2 * tid];
      const float* ga = xproj + (size_t)tn * HID4 + 2 * tid;
      asm volatile("global_load_async_to_lds_b64 %0, %1, off"
                   :: "v"(ldsa), "v"(ga) : "memory");
    }
    float a0 = xbuf[cur][2 * p * HID + u];
    float a1 = xbuf[cur][(2 * p + 1) * HID + u];
    for (int k = 0; k < HID; ++k) { float hk = h_sh[k]; a0 += w0[k] * hk; a1 += w1[k] * hk; }
    if (p == 1) { g_sh[0][u] = a0; g_sh[1][u] = a1; }
    __syncthreads();
    if (p == 0) {
      float ig = 1.f / (1.f + __expf(-a0));
      float fg = 1.f / (1.f + __expf(-a1));
      float gg = tanhf(g_sh[0][u]);
      float og = 1.f / (1.f + __expf(-g_sh[1][u]));
      c = fg * c + ig * gg;
      float hv = og * tanhf(c);
      h_sh[u] = hv;
      hout[(size_t)t * HID2 + dir * HID + u] = hv;
    }
    // ensure this wave's async stores to LDS have landed, then barrier so the
    // whole workgroup sees both h_sh and xbuf[nxt]
    asm volatile("s_wait_asynccnt 0x0" ::: "memory");
    __syncthreads();
  }
}

// ---------------- decoder kernels ----------------

// sproj[b][n] = s[b] . atts_w[n]   grid(H2,B), block 32
__global__ void k_dec_sproj(const float* __restrict__ s, const float* __restrict__ atts_w,
                            float* __restrict__ sproj) {
  int n = blockIdx.x, b = blockIdx.y, lane = threadIdx.x;
  const float* sr = s + b * HID;
  const float* wr = atts_w + (size_t)n * HID;
  float acc = 0.f;
  for (int k = lane; k < HID; k += 32) acc += sr[k] * wr[k];
  for (int off = 16; off; off >>= 1) acc += __shfl_down(acc, off, 32);
  if (lane == 0) sproj[b * HID2 + n] = acc;
}

// att[b][t] = dot(tanh(sproj[b]+hproj[t]), attw) masked   grid(T), block 128
__global__ void k_dec_att(const float* __restrict__ sproj, const float* __restrict__ hproj,
                          const float* __restrict__ attw, const int* __restrict__ lengthp,
                          float* __restrict__ att) {
  int t = blockIdx.x, tid = threadIdx.x;
  __shared__ float red[128];
  const float* hp = hproj + (size_t)t * HID2;
  bool valid = t < lengthp[0];
  float acc[NBEAM] = {0.f, 0.f, 0.f, 0.f};
  for (int k = tid; k < HID2; k += 128) {
    float hv = hp[k], wv = attw[k];
#pragma unroll
    for (int b = 0; b < NBEAM; ++b) acc[b] += tanhf(sproj[b * HID2 + k] + hv) * wv;
  }
  for (int b = 0; b < NBEAM; ++b) {
    red[tid] = acc[b]; __syncthreads();
    for (int off = 64; off; off >>= 1) { if (tid < off) red[tid] += red[tid + off]; __syncthreads(); }
    if (tid == 0) att[b * T_LEN + t] = valid ? red[0] : -1e30f;
    __syncthreads();
  }
}

// softmax over T per beam   grid(B), block 256
__global__ void k_dec_softmax(const float* __restrict__ att, float* __restrict__ alpha) {
  int b = blockIdx.x, tid = threadIdx.x;
  __shared__ float red[256];
  const float* ar = att + b * T_LEN;
  float m = -1e30f;
  for (int t = tid; t < T_LEN; t += 256) m = fmaxf(m, ar[t]);
  red[tid] = m; __syncthreads();
  for (int off = 128; off; off >>= 1) { if (tid < off) red[tid] = fmaxf(red[tid], red[tid + off]); __syncthreads(); }
  m = red[0]; __syncthreads();
  float sum = 0.f;
  for (int t = tid; t < T_LEN; t += 256) sum += __expf(ar[t] - m);
  red[tid] = sum; __syncthreads();
  for (int off = 128; off; off >>= 1) { if (tid < off) red[tid] += red[tid + off]; __syncthreads(); }
  float inv = 1.f / red[0];
  for (int t = tid; t < T_LEN; t += 256) alpha[b * T_LEN + t] = __expf(ar[t] - m) * inv;
}

// g[b][n] = sum_t alpha[b][t]*hpack[t][n]   grid(H2), block 128
__global__ void k_dec_g(const float* __restrict__ alpha, const float* __restrict__ hpack,
                        float* __restrict__ g) {
  int n = blockIdx.x, tid = threadIdx.x;
  __shared__ float red[128];
  float acc[NBEAM] = {0.f, 0.f, 0.f, 0.f};
  for (int t = tid; t < T_LEN; t += 128) {
    float hv = hpack[(size_t)t * HID2 + n];
#pragma unroll
    for (int b = 0; b < NBEAM; ++b) acc[b] += alpha[b * T_LEN + t] * hv;
  }
  for (int b = 0; b < NBEAM; ++b) {
    red[tid] = acc[b]; __syncthreads();
    for (int off = 64; off; off >>= 1) { if (tid < off) red[tid] += red[tid + off]; __syncthreads(); }
    if (tid == 0) g[b * HID2 + n] = red[0];
    __syncthreads();
  }
}

// pre16[b][n] = f16(tanh(s.sy_w[n] + g.gy_w[n] + gy_b[n]))   grid(H,B), block 32
__global__ void k_dec_pre(const float* __restrict__ s, const float* __restrict__ g,
                          const float* __restrict__ sy_w, const float* __restrict__ gy_w,
                          const float* __restrict__ gy_b, _Float16* __restrict__ pre16) {
  int n = blockIdx.x, b = blockIdx.y, lane = threadIdx.x;
  float acc = 0.f;
  const float* wr0 = sy_w + (size_t)n * HID;
  const float* wr1 = gy_w + (size_t)n * HID2;
  for (int k = lane; k < HID; k += 32)  acc += s[b * HID + k]  * wr0[k];
  for (int k = lane; k < HID2; k += 32) acc += g[b * HID2 + k] * wr1[k];
  for (int off = 16; off; off >>= 1) acc += __shfl_down(acc, off, 32);
  if (lane == 0) pre16[b * HID + n] = (_Float16)tanhf(acc + gy_b[n]);
}

// log-softmax over V per beam, emit f32 y and f16 y16 (ld VOCP)   grid(B), block 256
__global__ void k_dec_logsoftmax(const float* __restrict__ logits, float* __restrict__ y,
                                 _Float16* __restrict__ y16) {
  int b = blockIdx.x, tid = threadIdx.x;
  __shared__ float red[256];
  const float* lr = logits + (size_t)b * VOC;
  float m = -1e30f;
  for (int v = tid; v < VOC; v += 256) m = fmaxf(m, lr[v]);
  red[tid] = m; __syncthreads();
  for (int off = 128; off; off >>= 1) { if (tid < off) red[tid] = fmaxf(red[tid], red[tid + off]); __syncthreads(); }
  m = red[0]; __syncthreads();
  float sum = 0.f;
  for (int v = tid; v < VOC; v += 256) sum += __expf(lr[v] - m);
  red[tid] = sum; __syncthreads();
  for (int off = 128; off; off >>= 1) { if (tid < off) red[tid] += red[tid + off]; __syncthreads(); }
  float lse = m + logf(red[0]);
  for (int v = tid; v < VOC; v += 256) {
    float val = lr[v] - lse;
    y[b * VOC + v] = val;
    y16[(size_t)b * VOCP + v] = (_Float16)val;
  }
}

// gates[b][j] = tanh(s.ss_w[j] + g.gs_w[j] + gs_b[j] + gsum[b][j])   grid(H4,B), block 32
__global__ void k_dec_gates(const float* __restrict__ s, const float* __restrict__ g,
                            const float* __restrict__ ss_w, const float* __restrict__ gs_w,
                            const float* __restrict__ gs_b, const float* __restrict__ gsum,
                            float* __restrict__ gates) {
  int j = blockIdx.x, b = blockIdx.y, lane = threadIdx.x;
  float acc = 0.f;
  const float* wr0 = ss_w + (size_t)j * HID;
  const float* wr1 = gs_w + (size_t)j * HID2;
  for (int k = lane; k < HID; k += 32)  acc += s[b * HID + k]  * wr0[k];
  for (int k = lane; k < HID2; k += 32) acc += g[b * HID2 + k] * wr1[k];
  for (int off = 16; off; off >>= 1) acc += __shfl_down(acc, off, 32);
  if (lane == 0) gates[b * HID4 + j] = tanhf(acc + gs_b[j] + gsum[(size_t)b * HID4 + j]);
}

// cell update + per-beam top4 + global top4 + state/seq permutation   grid(1), block 1024
__global__ void k_dec_cell_topk(const float* __restrict__ gates, const float* __restrict__ c_in,
                                const float* __restrict__ y, const float* __restrict__ scores_in,
                                const int* __restrict__ seqs_in, float* __restrict__ c_out,
                                float* __restrict__ s_out, float* __restrict__ scores_out,
                                int* __restrict__ seqs_out, int t) {
  __shared__ float cn_sh[NBEAM][HID];
  __shared__ float hn_sh[NBEAM][HID];
  __shared__ float red_v[1024];
  __shared__ int   red_i[1024];
  __shared__ float topv_sh[NBEAM][NBEAM];
  __shared__ int   topi_sh[NBEAM][NBEAM];
  __shared__ int   parent_sh[NBEAM], tok_sh[NBEAM];
  __shared__ float ns_sh[NBEAM];
  int tid = threadIdx.x;

  for (int i = tid; i < NBEAM * HID; i += 1024) {
    int b = i / HID, u = i % HID;
    float gi = gates[b * HID4 + u];
    float gf = gates[b * HID4 + HID + u];
    float gc = gates[b * HID4 + 2 * HID + u];
    float go = gates[b * HID4 + 3 * HID + u];
    float ig = tanhf(gi * 0.5f) * 0.5f + 0.5f;
    float fg = tanhf(gf * 0.5f) * 0.5f + 0.5f;
    float og = tanhf(go * 0.5f) * 0.5f + 0.5f;
    float cn = fg * c_in[b * HID + u] + ig * tanhf(gc);
    cn_sh[b][u] = cn;
    hn_sh[b][u] = og * tanhf(cn);
  }
  __syncthreads();

  for (int b = 0; b < NBEAM; ++b) {
    for (int it = 0; it < NBEAM; ++it) {
      float best = -1e30f; int bi = 0;
      for (int v = tid; v < VOC; v += 1024) {
        bool skip = false;
        for (int q = 0; q < it; ++q) if (topi_sh[b][q] == v) skip = true;
        float val = y[b * VOC + v];
        if (!skip && val > best) { best = val; bi = v; }
      }
      red_v[tid] = best; red_i[tid] = bi; __syncthreads();
      for (int off = 512; off; off >>= 1) {
        if (tid < off && red_v[tid + off] > red_v[tid]) {
          red_v[tid] = red_v[tid + off]; red_i[tid] = red_i[tid + off];
        }
        __syncthreads();
      }
      if (tid == 0) { topv_sh[b][it] = red_v[0]; topi_sh[b][it] = red_i[0]; }
      __syncthreads();
    }
  }

  if (tid == 0) {
    float comb[16]; int used[16];
    for (int j = 0; j < 16; ++j) { comb[j] = scores_in[j / 4] + topv_sh[j / 4][j % 4]; used[j] = 0; }
    for (int nb = 0; nb < NBEAM; ++nb) {
      float best = -1e30f; int bj = 0;
      for (int j = 0; j < 16; ++j) if (!used[j] && comb[j] > best) { best = comb[j]; bj = j; }
      used[bj] = 1;
      parent_sh[nb] = bj / 4;
      tok_sh[nb] = topi_sh[bj / 4][bj % 4];
      ns_sh[nb] = best;
    }
  }
  __syncthreads();

  for (int i = tid; i < NBEAM * HID; i += 1024) {
    int b = i / HID, u = i % HID;
    int pb = parent_sh[b];
    c_out[b * HID + u] = cn_sh[pb][u];
    s_out[b * HID + u] = hn_sh[pb][u];
  }
  if (tid < NBEAM) scores_out[tid] = ns_sh[tid];
  if (tid < NBEAM * NSTEP) {
    int b = tid / NSTEP, j = tid % NSTEP;
    int pv = (j < t) ? seqs_in[parent_sh[b] * NSTEP + j] : 0;
    seqs_out[b * NSTEP + j] = (j == t) ? tok_sh[b] : pv;
  }
}

__global__ void k_dec_init(float* c, float* s, float* scores, _Float16* pre16) {
  int i = blockIdx.x * blockDim.x + threadIdx.x;
  if (i < NBEAM * HID) { c[i] = 0.f; s[i] = 0.f; }
  if (i < NBEAM) scores[i] = (i == 0) ? 0.f : -1e9f;
  if (i < 16 * HID) pre16[i] = (_Float16)0.f;
}

__global__ void k_finalize(const int* __restrict__ seqs, const float* __restrict__ scores,
                           float* __restrict__ out) {
  int i = blockIdx.x * blockDim.x + threadIdx.x;
  if (i < NBEAM * NSTEP) ((int*)out)[i] = seqs[i];
  else if (i < NBEAM * NSTEP + NBEAM) out[i] = scores[i - NBEAM * NSTEP];
}

// ---------------- host ----------------

static inline int cdiv_h(int a, int b) { return (a + b - 1) / b; }
static inline size_t alignup_h(size_t x) { return (x + 255) & ~(size_t)255; }

extern "C" void kernel_launch(void* const* d_in, const int* in_sizes, int n_in,
                              void* d_out, int out_size, void* d_ws, size_t ws_size,
                              hipStream_t stream) {
  (void)in_sizes; (void)n_in; (void)out_size; (void)ws_size;
  const float* data   = (const float*)d_in[0];
  const float* wih0   = (const float*)d_in[1];
  const float* whh0   = (const float*)d_in[2];
  const float* b0     = (const float*)d_in[3];
  const float* wihL   = (const float*)d_in[4];
  const float* whhL   = (const float*)d_in[5];
  const float* bL     = (const float*)d_in[6];
  const float* atts_w = (const float*)d_in[7];
  const float* atth_w = (const float*)d_in[8];
  const float* atth_b = (const float*)d_in[9];
  const float* attw_w = (const float*)d_in[10];
  const float* sy_w   = (const float*)d_in[11];
  const float* gy_w   = (const float*)d_in[12];
  const float* gy_b   = (const float*)d_in[13];
  const float* yy_w   = (const float*)d_in[14];
  const float* yy_b   = (const float*)d_in[15];
  const float* ss_w   = (const float*)d_in[16];
  const float* gs_w   = (const float*)d_in[17];
  const float* gs_b   = (const float*)d_in[18];
  const float* ys_w   = (const float*)d_in[19];
  const float* ys_b   = (const float*)d_in[20];
  const int*   lenp   = (const int*)d_in[21];

  char* wsb = (char*)d_ws;
  size_t off = 0;
  auto alloc = [&](size_t bytes) -> void* { void* p = wsb + off; off = alignup_h(off + bytes); return p; };

  _Float16* wih0T = (_Float16*)alloc((size_t)2 * DINP * HID4 * 2);
  _Float16* wihT  = (_Float16*)alloc((size_t)6 * HID2 * HID4 * 2);
  _Float16* atthT = (_Float16*)alloc((size_t)HID2 * HID2 * 2);
  _Float16* yyT   = (_Float16*)alloc((size_t)HID * VOC * 2);
  _Float16* ysT   = (_Float16*)alloc((size_t)VOCP * HID4 * 2);
  _Float16* X16   = (_Float16*)alloc((size_t)T_LEN * HID2 * 2);
  float* xprojF   = (float*)alloc((size_t)T_LEN * HID4 * 4);
  float* xprojB   = (float*)alloc((size_t)T_LEN * HID4 * 4);
  float* hA       = (float*)alloc((size_t)T_LEN * HID2 * 4);
  float* hB       = (float*)alloc((size_t)T_LEN * HID2 * 4);
  _Float16* hpack16 = (_Float16*)alloc((size_t)T_LEN * HID2 * 2);
  float* hproj    = (float*)alloc((size_t)T_LEN * HID2 * 4);
  float* sproj    = (float*)alloc((size_t)NBEAM * HID2 * 4);
  float* att      = (float*)alloc((size_t)NBEAM * T_LEN * 4);
  float* alpha    = (float*)alloc((size_t)NBEAM * T_LEN * 4);
  float* gctx     = (float*)alloc((size_t)NBEAM * HID2 * 4);
  _Float16* pre16 = (_Float16*)alloc((size_t)16 * HID * 2);
  float* logits   = (float*)alloc((size_t)16 * VOC * 4);
  float* ybuf     = (float*)alloc((size_t)NBEAM * VOC * 4);
  _Float16* y16   = (_Float16*)alloc((size_t)16 * VOCP * 2);
  float* gsum     = (float*)alloc((size_t)16 * HID4 * 4);
  float* gates    = (float*)alloc((size_t)NBEAM * HID4 * 4);
  float* cA = (float*)alloc(NBEAM * HID * 4);
  float* cB = (float*)alloc(NBEAM * HID * 4);
  float* sA = (float*)alloc(NBEAM * HID * 4);
  float* sB = (float*)alloc(NBEAM * HID * 4);
  float* scA = (float*)alloc(256);
  float* scB = (float*)alloc(256);
  int* seqA = (int*)alloc(NBEAM * NSTEP * 4);
  int* seqB = (int*)alloc(NBEAM * NSTEP * 4);

  // ---- per-call weight conversion: K-major f16 copies ----
  for (int d = 0; d < 2; ++d)
    k_transpose_f16<<<cdiv_h(DINP * HID4, 256), 256, 0, stream>>>(
        wih0 + (size_t)d * HID4 * DIN_, wih0T + (size_t)d * DINP * HID4, HID4, DIN_, DINP);
  for (int ld = 0; ld < 6; ++ld)
    k_transpose_f16<<<cdiv_h(HID2 * HID4, 256), 256, 0, stream>>>(
        wihL + (size_t)ld * HID4 * HID2, wihT + (size_t)ld * HID2 * HID4, HID4, HID2, HID2);
  k_transpose_f16<<<cdiv_h(HID2 * HID2, 256), 256, 0, stream>>>(atth_w, atthT, HID2, HID2, HID2);
  k_transpose_f16<<<cdiv_h(HID * VOC, 256), 256, 0, stream>>>(yy_w, yyT, VOC, HID, HID);
  k_transpose_f16<<<cdiv_h(VOCP * HID4, 256), 256, 0, stream>>>(ys_w, ysT, HID4, VOC, VOCP);

  // ---- encoder: 4 BiLSTM layers ----
  const float* layer_in = data;
  float* hcur = hA;
  int K = DIN_, Kp = DINP;
  const float* hpack = nullptr;
  for (int l = 0; l < 4; ++l) {
    k_f32_to_f16_pad<<<cdiv_h(T_LEN * Kp, 256), 256, 0, stream>>>(layer_in, X16, T_LEN, K, Kp);
    int ngroups = (HID4 / 16) / 4;                 // 20 groups of 4 N-tiles
    int waves = (T_LEN / 16) * ngroups;            // 2500
    for (int d = 0; d < 2; ++d) {
      const _Float16* Bt = (l == 0) ? wih0T + (size_t)d * DINP * HID4
                                    : wihT + (size_t)((l - 1) * 2 + d) * HID2 * HID4;
      const float* bias = (l == 0) ? b0 + (size_t)d * HID4
                                   : bL + (size_t)((l - 1) * 2 + d) * HID4;
      k_wmma_gemm_nb<4><<<cdiv_h(waves, 8), 256, 0, stream>>>(
          X16, Bt, bias, d ? xprojB : xprojF, T_LEN, HID4, Kp, ngroups, 0);
    }
    const float* whhp = (l == 0) ? whh0 : whhL + (size_t)(l - 1) * 2 * HID4 * HID;
    k_lstm_scan<<<2, 640, 0, stream>>>(xprojF, xprojB, whhp, hcur);
    layer_in = hcur;
    hpack = hcur;
    hcur = (hcur == hA) ? hB : hA;
    K = HID2; Kp = HID2;
  }

  // ---- attention precompute: hproj = hpack @ atth_w^T + atth_b ----
  k_f32_to_f16_pad<<<cdiv_h(T_LEN * HID2, 256), 256, 0, stream>>>(hpack, hpack16, T_LEN, HID2, HID2);
  {
    int ngroups = (HID2 / 16) / 4;                 // 10
    int waves = (T_LEN / 16) * ngroups;            // 1250
    k_wmma_gemm_nb<4><<<cdiv_h(waves, 8), 256, 0, stream>>>(
        hpack16, atthT, atth_b, hproj, T_LEN, HID2, HID2, ngroups, 0);
  }

  // ---- decoder init ----
  k_dec_init<<<cdiv_h(16 * HID, 256), 256, 0, stream>>>(cA, sA, scA, pre16);
  k_zero_f16<<<cdiv_h(16 * VOCP, 256), 256, 0, stream>>>(y16, 16 * VOCP);

  float *c_cur = cA, *c_nxt = cB, *s_cur = sA, *s_nxt = sB;
  float *sc_cur = scA, *sc_nxt = scB;
  int *sq_cur = seqA, *sq_nxt = seqB;

  for (int t = 0; t < NSTEP; ++t) {
    k_dec_sproj<<<dim3(HID2, NBEAM), 32, 0, stream>>>(s_cur, atts_w, sproj);
    k_dec_att<<<T_LEN, 128, 0, stream>>>(sproj, hproj, attw_w, lenp, att);
    k_dec_softmax<<<NBEAM, 256, 0, stream>>>(att, alpha);
    k_dec_g<<<HID2, 128, 0, stream>>>(alpha, hpack, gctx);
    k_dec_pre<<<dim3(HID, NBEAM), 32, 0, stream>>>(s_cur, gctx, sy_w, gy_w, gy_b, pre16);
    // logits(16 x 10000) = pre16 @ yyT + yy_b   (WMMA, NB=5: 625 tiles = 125 groups)
    k_wmma_gemm_nb<5><<<cdiv_h(125, 8), 256, 0, stream>>>(pre16, yyT, yy_b, logits, 16, VOC, HID, 125, 0);
    k_dec_logsoftmax<<<NBEAM, 256, 0, stream>>>(logits, ybuf, y16);
    // gsum(16 x 1280) = y16 @ ysT + ys_b   (WMMA, K = 10016, NB=4: 20 groups)
    k_wmma_gemm_nb<4><<<cdiv_h(20, 8), 256, 0, stream>>>(y16, ysT, ys_b, gsum, 16, HID4, VOCP, 20, 0);
    k_dec_gates<<<dim3(HID4, NBEAM), 32, 0, stream>>>(s_cur, gctx, ss_w, gs_w, gs_b, gsum, gates);
    k_dec_cell_topk<<<1, 1024, 0, stream>>>(gates, c_cur, ybuf, sc_cur, sq_cur,
                                            c_nxt, s_nxt, sc_nxt, sq_nxt, t);
    float* tf;
    tf = c_cur; c_cur = c_nxt; c_nxt = tf;
    tf = s_cur; s_cur = s_nxt; s_nxt = tf;
    tf = sc_cur; sc_cur = sc_nxt; sc_nxt = tf;
    int* ti = sq_cur; sq_cur = sq_nxt; sq_nxt = ti;
  }

  k_finalize<<<cdiv_h(NBEAM * NSTEP + NBEAM, 256), 256, 0, stream>>>(sq_cur, sc_cur, (float*)d_out);
}
